// Siamese_Model_2l_LSTM_25022479466790
// MI455X (gfx1250) — compile-verified, matrix operand order checked by reading, not run on hardware
//
#include <hip/hip_runtime.h>
#include <hip/hip_bf16.h>
#include <math.h>

typedef __attribute__((ext_vector_type(16))) _Float16 v16h;
typedef __attribute__((ext_vector_type(8)))  _Float16 v8h;
typedef __attribute__((ext_vector_type(8)))  float    v8f;

#define BATCH  1024
#define TSTEPS 128
#define EDIM   128
#define HDIM   256
#define GDIM   1024   /* 4*H */
#define NTILES 64     /* GDIM/16 */
#define EPSF   1e-12f

__device__ __forceinline__ float sigf(float x){ return 1.0f / (1.0f + __expf(-x)); }

__device__ __forceinline__ v8f vsplat(float x){
  v8f r;
#pragma unroll
  for (int j = 0; j < 8; ++j) r[j] = x;
  return r;
}

// A-fragment (16x32 f16) per lane: row m = lane&15; halves 0..7 at k = kt*32 + g*8,
// halves 8..15 at k = kt*32 + 16 + g*8  (g = lane>>4). Two 16B LDS loads.
__device__ __forceinline__ v16h load_afrag(const _Float16* p){
  v8h lo = *(const v8h*)p;
  v8h hi = *(const v8h*)(p + 16);
  return __builtin_shufflevector(lo, hi, 0,1,2,3,4,5,6,7,8,9,10,11,12,13,14,15);
}

// -------- weight pre-pack: W[K,N] f32 -> fragment-linear f16 --------
// packed[((kt*NT + nt)*32 + lane)*16 + j] = W[k, nt*16 + (lane&15)]
// with k = kt*32 + ((j<8)?0:16) + (lane>>4)*8 + (j&7)
__global__ __launch_bounds__(256) void pack_weights(const float* __restrict__ W,
                                                    _Float16* __restrict__ P,
                                                    int K, int N){
  int idx = blockIdx.x * 256 + threadIdx.x;
  if (idx >= K * N) return;
  int j    = idx & 15;
  int lane = (idx >> 4) & 31;
  int tile = idx >> 9;              // kt*NT + nt
  int NT   = N >> 4;
  int nt   = tile % NT;
  int kt   = tile / NT;
  int g    = lane >> 4;
  int n    = lane & 15;
  int k    = kt * 32 + ((j < 8) ? 0 : 16) + g * 8 + (j & 7);
  P[idx] = (_Float16)W[(size_t)k * N + nt * 16 + n];
}

// -------- GEMM helper: acc[gate] += A(src) * B(pw) over ktiles --------
// Wave `wave` (0..15) owns 16 output columns: col = gate*256 + wave*16 + n.
// An opaque zero offset (laundered via empty asm) defeats LICM on the t-invariant
// B-fragment loads WITHOUT stripping the pointer's global address space, so the
// loads stay global_load_b128 (LOADcnt only, decoupled from DScnt).
__device__ __forceinline__ void gemm_block(const _Float16* __restrict__ src, int rowStride,
                                           int ktiles, const _Float16* __restrict__ pw,
                                           int wave, int lane, v8f acc[4]){
  size_t off = 0;
  asm volatile("" : "+s"(off));        // opaque 0: loads depend on it -> stay in-loop
  const _Float16* __restrict__ p = pw + off;
  const int g = lane >> 4, m = lane & 15;
#pragma unroll
  for (int kt = 0; kt < ktiles; ++kt){
    v16h a = load_afrag(src + m * rowStride + kt * 32 + g * 8);
    const _Float16* tb = p + (size_t)kt * NTILES * 512;   // 512 halves per 32x16 tile
#pragma unroll
    for (int gate = 0; gate < 4; ++gate){
      int nt = gate * 16 + wave;
      v16h bf = *(const v16h*)(tb + (nt * 32 + lane) * 16);
      acc[gate] = __builtin_amdgcn_wmma_f32_16x16x32_f16(
          false, a, false, bf, (short)0, acc[gate], false, false);
    }
  }
}

// -------- gate nonlinearities + masked cell/hidden update (registers) --------
__device__ __forceinline__ void cell_update(const v8f acc[4], v8f& cst,
                                            const _Float16 (*hs)[264],
                                            const float* __restrict__ msk,
                                            int wave, int lane, float hn[8]){
  const int g = lane >> 4, nlo = lane & 15;
  const int col = wave * 16 + nlo;
#pragma unroll
  for (int r = 0; r < 8; ++r){
    const int m = r + g * 8;
    float iv = sigf(acc[0][r]);
    float fv = sigf(acc[1][r]);
    float gv = tanhf(acc[2][r]);
    float ov = sigf(acc[3][r]);
    float cold = cst[r];
    float cn   = fv * cold + iv * gv;
    float hnv  = ov * tanhf(cn);
    float mm   = msk[m];
    float hold = (float)hs[m][col];
    cst[r] = mm * cn  + (1.0f - mm) * cold;
    hn[r]  = mm * hnv + (1.0f - mm) * hold;
  }
}

__device__ __forceinline__ void store_h(_Float16 (*hs)[264], int wave, int lane,
                                        const float hn[8]){
  const int g = lane >> 4, nlo = lane & 15;
#pragma unroll
  for (int r = 0; r < 8; ++r)
    hs[r + g * 8][wave * 16 + nlo] = (_Float16)hn[r];
}

// -------- fused 2-layer LSTM, persistent per 16-row batch slice --------
// 512 threads = 16 waves; wave w owns 16 h-columns (all 4 gates) -> low VGPR pressure.
__global__ __launch_bounds__(512) void lstm_fused(
    const int* __restrict__ ids1, const int* __restrict__ ids2,
    const float* __restrict__ emb,
    const float* __restrict__ b1, const float* __restrict__ b2,
    const _Float16* __restrict__ pW1, const _Float16* __restrict__ pU1,
    const _Float16* __restrict__ pW2, const _Float16* __restrict__ pU2,
    float* __restrict__ outc){
  __shared__ _Float16 hx [16][136];   // x_t as f16, row pad -> conflict-free ds reads
  __shared__ _Float16 h1s[16][264];   // layer-1 h (f16)
  __shared__ _Float16 h2s[16][264];   // layer-2 h (f16)
  __shared__ float    msk[16];

  const int wgid   = blockIdx.x;
  const int branch = wgid >> 6;
  const int row0   = (wgid & 63) * 16;
  const int* __restrict__ ids = branch ? ids2 : ids1;

  const int tid = threadIdx.x, wave = tid >> 5, lane = tid & 31;
  const int g = lane >> 4, nlo = lane & 15;

  for (int i = tid; i < 16 * 264; i += 512){
    (&h1s[0][0])[i] = (_Float16)0.0f;
    (&h2s[0][0])[i] = (_Float16)0.0f;
  }

  float bb1[4], bb2[4];
#pragma unroll
  for (int gate = 0; gate < 4; ++gate){
    int col = gate * 256 + wave * 16 + nlo;
    bb1[gate] = b1[col];
    bb2[gate] = b2[col];
  }

  v8f c1st = vsplat(0.0f);
  v8f c2st = vsplat(0.0f);

  const int rG = tid >> 5;          // gather row 0..15
  const int cG = (tid & 31) * 4;    // gather col chunk (one float4)

#pragma unroll 1
  for (int t = 0; t < TSTEPS; ++t){
    // ---- mask + embedding gather (f32 -> f16 into LDS) ----
    if (tid < 16) msk[tid] = (ids[(row0 + tid) * TSTEPS + t] != 0) ? 1.0f : 0.0f;
    {
      int id = ids[(row0 + rG) * TSTEPS + t];
      const float4* ep = (const float4*)(emb + (size_t)id * EDIM + cG);
      float4 v0 = ep[0];
      _Float16* dst = &hx[rG][cG];
      dst[0] = (_Float16)v0.x; dst[1] = (_Float16)v0.y;
      dst[2] = (_Float16)v0.z; dst[3] = (_Float16)v0.w;
    }
    __syncthreads();

    // ---- layer 1: z = x@W1 + h1@U1 + b1 ----
    {
      v8f acc[4];
#pragma unroll
      for (int gate = 0; gate < 4; ++gate) acc[gate] = vsplat(bb1[gate]);
      gemm_block(&hx [0][0], 136, 4, pW1, wave, lane, acc);  // K = 128
      gemm_block(&h1s[0][0], 264, 8, pU1, wave, lane, acc);  // K = 256 (h_{t-1})
      float hn[8];
      cell_update(acc, c1st, h1s, msk, wave, lane, hn);
      __syncthreads();                 // all waves done reading old h1
      store_h(h1s, wave, lane, hn);
    }
    __syncthreads();

    // ---- layer 2: z = h1@W2 + h2@U2 + b2 ----
    {
      v8f acc[4];
#pragma unroll
      for (int gate = 0; gate < 4; ++gate) acc[gate] = vsplat(bb2[gate]);
      gemm_block(&h1s[0][0], 264, 8, pW2, wave, lane, acc);
      gemm_block(&h2s[0][0], 264, 8, pU2, wave, lane, acc);
      float hn[8];
      cell_update(acc, c2st, h2s, msk, wave, lane, hn);
      __syncthreads();                 // all waves done reading old h2
      store_h(h2s, wave, lane, hn);
    }
    __syncthreads();
  }

  // ---- final layer-2 cell state -> d_out (c1 region for branch 0, c2 for branch 1)
  float* co = outc + (size_t)branch * BATCH * HDIM;
#pragma unroll
  for (int r = 0; r < 8; ++r){
    int m = r + g * 8;
    co[(size_t)(row0 + m) * HDIM + wave * 16 + nlo] = c2st[r];
  }
}

// -------- cosine similarity: one wave per batch row --------
__global__ __launch_bounds__(256) void sim_kernel(float* __restrict__ out){
  int b    = blockIdx.x * 8 + (threadIdx.x >> 5);
  int lane = threadIdx.x & 31;
  const float* c1 = out + (size_t)b * HDIM;
  const float* c2 = out + (size_t)BATCH * HDIM + (size_t)b * HDIM;
  float s11 = 0.0f, s22 = 0.0f, s12 = 0.0f;
#pragma unroll
  for (int j = 0; j < 8; ++j){
    float a = c1[j * 32 + lane];
    float c = c2[j * 32 + lane];
    s11 += a * a; s22 += c * c; s12 += a * c;
  }
#pragma unroll
  for (int off = 16; off > 0; off >>= 1){
    s11 += __shfl_xor(s11, off, 32);
    s22 += __shfl_xor(s22, off, 32);
    s12 += __shfl_xor(s12, off, 32);
  }
  if (lane == 0)
    out[(size_t)2 * BATCH * HDIM + b] =
        s12 * rsqrtf(fmaxf(s11, EPSF)) * rsqrtf(fmaxf(s22, EPSF));
}

extern "C" void kernel_launch(void* const* d_in, const int* in_sizes, int n_in,
                              void* d_out, int out_size, void* d_ws, size_t ws_size,
                              hipStream_t stream){
  (void)in_sizes; (void)n_in; (void)out_size; (void)ws_size;
  const int*   ids1 = (const int*)  d_in[0];
  const int*   ids2 = (const int*)  d_in[1];
  const float* emb  = (const float*)d_in[2];
  const float* W1   = (const float*)d_in[3];
  const float* U1   = (const float*)d_in[4];
  const float* b1   = (const float*)d_in[5];
  const float* W2   = (const float*)d_in[6];
  const float* U2   = (const float*)d_in[7];
  const float* b2   = (const float*)d_in[8];
  float* out = (float*)d_out;

  _Float16* ws  = (_Float16*)d_ws;
  _Float16* pW1 = ws;
  _Float16* pU1 = pW1 + (size_t)128 * 1024;
  _Float16* pW2 = pU1 + (size_t)256 * 1024;
  _Float16* pU2 = pW2 + (size_t)256 * 1024;

  pack_weights<<<(128 * 1024) / 256, 256, 0, stream>>>(W1, pW1, 128, 1024);
  pack_weights<<<(256 * 1024) / 256, 256, 0, stream>>>(U1, pU1, 256, 1024);
  pack_weights<<<(256 * 1024) / 256, 256, 0, stream>>>(W2, pW2, 256, 1024);
  pack_weights<<<(256 * 1024) / 256, 256, 0, stream>>>(U2, pU2, 256, 1024);

  lstm_fused<<<128, 512, 0, stream>>>(ids1, ids2, emb, b1, b2,
                                      pW1, pU1, pW2, pU2, out);
  sim_kernel<<<128, 256, 0, stream>>>(out);
}
